// SelfAttention_19653770346998
// MI455X (gfx1250) — matrix operand order repaired and checked by reading, not compiled, not run on hardware
//
#include <hip/hip_runtime.h>

// ---------------------------------------------------------------------------
// SAGAN self-attention, fused flash-attention style, bf16 WMMA (gfx1250)
// B=4, H=W=64 -> N=4096, C=64, Ck=8 (padded to 32 for the K=32 bf16 WMMA)
// Round 2: BK=64, row-sums via P@ones WMMA, max-reduce via v_permlane16,
//          software-pipelined key-tile loads.
// ---------------------------------------------------------------------------

typedef __attribute__((ext_vector_type(16))) __bf16       v16bf;
typedef __attribute__((ext_vector_type(8)))  float        v8f;
typedef __attribute__((ext_vector_type(4)))  unsigned int v4u;

#define B_   4
#define N_   4096
#define C_   64
#define CK_  8
#define CKP_ 32      // padded key/query channels (WMMA K)
#define MQ_  128     // query rows per workgroup (8 waves x 16)
#define BK_  64      // keys per inner iteration (two K=32 steps for PV)
#define NBK_ (N_ / BK_)

union ABOp { v16bf v; v4u u[2]; };

__device__ __forceinline__ v8f wmma_bf16(v16bf a, v16bf b, v8f c) {
  return __builtin_amdgcn_wmma_f32_16x16x32_bf16(false, a, false, b,
                                                 (short)0, c, false, false);
}
__device__ __forceinline__ void wait_ds() {
  asm volatile("s_wait_dscnt 0" ::: "memory");
}
// lane i reads lane (i ^ m) within its 16-lane row; VALU pipe, no LDS traffic
__device__ __forceinline__ float pl16(float x, unsigned s0, unsigned s1) {
  int xi = __builtin_bit_cast(int, x);
  int r  = __builtin_amdgcn_permlane16(xi, xi, s0, s1, false, false);
  return __builtin_bit_cast(float, r);
}
__device__ __forceinline__ float rowmax16(float t) {
  t = fmaxf(t, pl16(t, 0x67452301u, 0xEFCDAB89u));   // xor 1
  t = fmaxf(t, pl16(t, 0x54761032u, 0xDCFE98BAu));   // xor 2
  t = fmaxf(t, pl16(t, 0x32107654u, 0xBA98FEDCu));   // xor 4
  t = fmaxf(t, pl16(t, 0xFEDCBA98u, 0x76543210u));   // xor 8
  return t;
}

// ---------------------------------------------------------------------------
// Weight prep: WoT[cout][cin] = bf16(wo[cin][cout])
// ---------------------------------------------------------------------------
__global__ void prep_wo(const float* __restrict__ wo, __bf16* __restrict__ WoT) {
  int i = blockIdx.x * blockDim.x + threadIdx.x;   // 0..4095
  int cout = i >> 6, cin = i & 63;
  WoT[cout * C_ + cin] = (__bf16)wo[cin * C_ + cout];
}

// ---------------------------------------------------------------------------
// Projections: f = x@wf+bf (keys), g = x@wg+bg (queries), h = x@wh+bh (values)
// Fp/Qp: [B][N][32] bf16 zero-padded;  Ht: [B][C][N] bf16 (transposed values)
// ---------------------------------------------------------------------------
__global__ __launch_bounds__(64)
void proj_kernel(const float* __restrict__ x,
                 const float* __restrict__ wf, const float* __restrict__ bfp,
                 const float* __restrict__ wg, const float* __restrict__ bgp,
                 const float* __restrict__ wh, const float* __restrict__ bhp,
                 __bf16* __restrict__ Qp, __bf16* __restrict__ Fp,
                 __bf16* __restrict__ Ht) {
  __shared__ float xr[C_];
  const int p = blockIdx.x;            // pixel over B*N
  const int b = p / N_;
  const int n = p % N_;
  const int t = threadIdx.x;
  xr[t] = x[(size_t)p * C_ + t];
  __syncthreads();

  float acc = bhp[t];                  // value projection (64 -> 64)
  #pragma unroll
  for (int c = 0; c < C_; ++c) acc += xr[c] * wh[c * C_ + t];
  Ht[((size_t)b * C_ + t) * N_ + n] = (__bf16)acc;

  if (t < CKP_) {                      // key/query projections (64 -> 8, padded)
    float fv = 0.f, gv = 0.f;
    if (t < CK_) {
      fv = bfp[t]; gv = bgp[t];
      #pragma unroll
      for (int c = 0; c < C_; ++c) {
        fv += xr[c] * wf[c * CK_ + t];
        gv += xr[c] * wg[c * CK_ + t];
      }
    }
    Fp[(size_t)p * CKP_ + t] = (__bf16)fv;
    Qp[(size_t)p * CKP_ + t] = (__bf16)gv;
  }
}

// ---------------------------------------------------------------------------
// Fused attention: per wave, 16 query rows; online softmax over 64-key blocks
// ---------------------------------------------------------------------------
__global__ __launch_bounds__(256)
void flash_kernel(const __bf16* __restrict__ Qp, const __bf16* __restrict__ Fp,
                  const __bf16* __restrict__ Ht, const __bf16* __restrict__ WoT,
                  const float* __restrict__ bo, const float* __restrict__ gamma,
                  const float* __restrict__ x, float* __restrict__ out) {
  __shared__ __bf16 smem[8 * 16 * C_];                 // 2 KB per wave
  const int wave = threadIdx.x >> 5;
  const int lane = threadIdx.x & 31;
  const int lrow = lane & 15;
  const int hi   = lane >> 4;
  const int b    = blockIdx.x / (N_ / MQ_);
  const int qt   = blockIdx.x % (N_ / MQ_);
  const int q0   = qt * MQ_ + wave * 16;

  __bf16* lds_w = smem + wave * (16 * C_);

  // Q as WMMA A operand (16x32 bf16): lanes<16 -> k{0..15}, lanes>=16 -> k{16..31}
  ABOp qa;
  {
    const __bf16* qrow = Qp + ((size_t)b * N_ + q0 + lrow) * CKP_;
    qa.u[0] = *(const v4u*)(qrow + (hi ? 16 : 0));
    qa.u[1] = *(const v4u*)(qrow + (hi ? 24 : 8));
  }

  // all-ones B operand (bf16 1.0 splat) for row sums via WMMA
  ABOp ones;
  ones.u[0] = (v4u){0x3F803F80u, 0x3F803F80u, 0x3F803F80u, 0x3F803F80u};
  ones.u[1] = ones.u[0];

  const v8f vzero = {0.f, 0.f, 0.f, 0.f, 0.f, 0.f, 0.f, 0.f};
  v8f o_acc[4] = {vzero, vzero, vzero, vzero};         // 16 rows x 64 channels
  v8f l_acc    = vzero;                                // running row sums
  float m_row[8];
  #pragma unroll
  for (int v = 0; v < 8; ++v) m_row[v] = -1e30f;

  const __bf16* Fb = Fp + (size_t)b * N_ * CKP_;

  // preload key tiles for block 0 (B 32x16 layout: lane<16 -> n=lane,k=0..15;
  // lane>=16 -> n=lane-16,k=16..31)
  ABOp fb[4];
  #pragma unroll
  for (int j = 0; j < 4; ++j) {
    const __bf16* fr = Fb + (size_t)(j * 16 + lrow) * CKP_ + hi * 16;
    fb[j].u[0] = *(const v4u*)(fr);
    fb[j].u[1] = *(const v4u*)(fr + 8);
  }

  for (int kb = 0; kb < NBK_; ++kb) {
    const int key0 = kb * BK_;

    v8f s[4];
    #pragma unroll
    for (int j = 0; j < 4; ++j) s[j] = wmma_bf16(qa.v, fb[j].v, vzero);

    // software pipeline: issue next block's key-tile loads now (latency hidden
    // behind the softmax VALU work below)
    const int keyn = (kb + 1 < NBK_) ? (kb + 1) * BK_ : 0;
    #pragma unroll
    for (int j = 0; j < 4; ++j) {
      const __bf16* fr = Fb + (size_t)(keyn + j * 16 + lrow) * CKP_ + hi * 16;
      fb[j].u[0] = *(const v4u*)(fr);
      fb[j].u[1] = *(const v4u*)(fr + 8);
    }

    // Online softmax. D layout: VGPR v, lane -> (m = v + 8*hi, n = lane&15).
    #pragma unroll
    for (int v = 0; v < 8; ++v) {
      float t = fmaxf(fmaxf(s[0][v], s[1][v]), fmaxf(s[2][v], s[3][v]));
      t = rowmax16(t);
      const float mnew = fmaxf(m_row[v], t);
      const float corr = __expf(m_row[v] - mnew);
      m_row[v] = mnew;
      const int m = v + 8 * hi;
      #pragma unroll
      for (int j = 0; j < 4; ++j)
        lds_w[m * BK_ + j * 16 + lrow] = (__bf16)__expf(s[j][v] - mnew);
      #pragma unroll
      for (int nt = 0; nt < 4; ++nt) o_acc[nt][v] *= corr;
      l_acc[v] *= corr;
    }
    wait_ds();                                         // cross-lane LDS visibility

    // Re-load P in A layout (two 16x32 K-steps) and accumulate P@V and P@ones
    #pragma unroll
    for (int ks = 0; ks < 2; ++ks) {
      ABOp pa;
      const __bf16* pr = lds_w + lrow * BK_ + ks * 32 + hi * 16;
      pa.u[0] = *(const v4u*)(pr);
      pa.u[1] = *(const v4u*)(pr + 8);
      #pragma unroll
      for (int nt = 0; nt < 4; ++nt) {
        const __bf16* vrow =
            Ht + ((size_t)b * C_ + nt * 16 + lrow) * N_ + key0 + ks * 32 + hi * 16;
        ABOp vb;
        vb.u[0] = *(const v4u*)(vrow);
        vb.u[1] = *(const v4u*)(vrow + 8);
        o_acc[nt] = wmma_bf16(pa.v, vb.v, o_acc[nt]);
      }
      l_acc = wmma_bf16(pa.v, ones.v, l_acc);          // row sums, replicated
    }
  }

  // ---- epilogue: normalize, o @ wo via WMMA, out = gamma*(..+bo) + x ----
  #pragma unroll
  for (int v = 0; v < 8; ++v) {
    const int m = v + 8 * hi;
    const float inv = 1.0f / l_acc[v];
    #pragma unroll
    for (int nt = 0; nt < 4; ++nt)
      lds_w[m * C_ + nt * 16 + lrow] = (__bf16)(o_acc[nt][v] * inv);
  }
  wait_ds();

  v8f acc2[4] = {vzero, vzero, vzero, vzero};
  #pragma unroll
  for (int ks = 0; ks < 2; ++ks) {                     // K = 64 in two 32-steps
    ABOp oa;
    const __bf16* orow = lds_w + lrow * C_ + ks * 32 + hi * 16;
    oa.u[0] = *(const v4u*)(orow);
    oa.u[1] = *(const v4u*)(orow + 8);
    #pragma unroll
    for (int nt = 0; nt < 4; ++nt) {
      const __bf16* wrow = WoT + (nt * 16 + lrow) * C_ + ks * 32 + hi * 16;
      ABOp wb;
      wb.u[0] = *(const v4u*)(wrow);
      wb.u[1] = *(const v4u*)(wrow + 8);
      acc2[nt] = wmma_bf16(oa.v, wb.v, acc2[nt]);
    }
  }

  const float g0 = gamma[0];
  #pragma unroll
  for (int nt = 0; nt < 4; ++nt) {
    const int ch = nt * 16 + lrow;
    const float bov = bo[ch];
    #pragma unroll
    for (int v = 0; v < 8; ++v) {
      const int row = q0 + v + 8 * hi;
      const size_t idx = ((size_t)b * N_ + row) * C_ + ch;
      out[idx] = g0 * (acc2[nt][v] + bov) + x[idx];
    }
  }
}

// ---------------------------------------------------------------------------
extern "C" void kernel_launch(void* const* d_in, const int* in_sizes, int n_in,
                              void* d_out, int out_size, void* d_ws, size_t ws_size,
                              hipStream_t stream) {
  const float* x     = (const float*)d_in[0];
  const float* wf    = (const float*)d_in[1];
  const float* bfp   = (const float*)d_in[2];
  const float* wg    = (const float*)d_in[3];
  const float* bgp   = (const float*)d_in[4];
  const float* wh    = (const float*)d_in[5];
  const float* bhp   = (const float*)d_in[6];
  const float* wo    = (const float*)d_in[7];
  const float* bo    = (const float*)d_in[8];
  const float* gamma = (const float*)d_in[9];
  float* out = (float*)d_out;

  char* ws = (char*)d_ws;
  const size_t qp_bytes = (size_t)B_ * N_ * CKP_ * 2;       // 1 MB
  __bf16* Qp  = (__bf16*)ws;  ws += qp_bytes;
  __bf16* Fp  = (__bf16*)ws;  ws += qp_bytes;
  __bf16* Ht  = (__bf16*)ws;  ws += (size_t)B_ * C_ * N_ * 2;  // 2 MB
  __bf16* WoT = (__bf16*)ws;                                   // 8 KB

  prep_wo   <<<dim3(16),          dim3(256), 0, stream>>>(wo, WoT);
  proj_kernel<<<dim3(B_ * N_),    dim3(64),  0, stream>>>(x, wf, bfp, wg, bgp,
                                                          wh, bhp, Qp, Fp, Ht);
  flash_kernel<<<dim3(B_ * (N_ / MQ_)), dim3(256), 0, stream>>>(
      Qp, Fp, Ht, WoT, bo, gamma, x, out);
}
